// GATConvolution_24953759990542
// MI455X (gfx1250) — matrix operand order repaired and checked by reading, not compiled, not run on hardware
//
#include <hip/hip_runtime.h>
#include <hip/hip_bf16.h>
#include <math.h>

typedef __bf16 bf16;
typedef __attribute__((ext_vector_type(16))) __bf16 v16bf;
typedef __attribute__((ext_vector_type(8)))  float  v8f;

#define NB   4
#define NN   4096
#define DIN  512
#define DOUT 128
#define NEG_INF -1000000000.0f

// ---------------------------------------------------------------------------
// WMMA helper: D(16x16 f32) = A(16x32 bf16) * B(32x16 bf16) + C
// ---------------------------------------------------------------------------
__device__ __forceinline__ v8f wmma_bf16(v16bf a, v16bf b, v8f c) {
  return __builtin_amdgcn_wmma_f32_16x16x32_bf16(
      /*neg_a=*/false, a, /*neg_b=*/false, b,
      /*c_mod=*/(short)0, c, /*reuse_a=*/false, /*reuse_b=*/false);
}

// A-fragment (16x32, bf16) loaded from a row-major f32 matrix with leading dim ld.
// ISA layout: lanes 0-15 hold row M=lane, K runs {0-7,16-23}; lanes 16-31 hold
// row M=lane-16, K runs {8-15,24-31}; 2 bf16 per VGPR across 8 VGPRs.
__device__ __forceinline__ v16bf load_a_f32(const float* __restrict__ p0, int ld, int lane) {
  const float* p = p0 + (lane & 15) * ld + ((lane & 16) ? 8 : 0);
  v16bf a;
#pragma unroll
  for (int t = 0; t < 8; ++t) a[t] = (bf16)p[t];
#pragma unroll
  for (int t = 0; t < 8; ++t) a[8 + t] = (bf16)p[16 + t];
  return a;
}

// B-fragment (32x16, bf16) loaded from a TRANSPOSED bf16 operand stored [N][K]
// with leading dim ld. ISA layout: lanes 0-15 = column N=lane, K runs {0-7,16-23};
// lanes 16-31 = column N=lane-16, K runs {8-15,24-31}.
__device__ __forceinline__ v16bf load_b_T(const bf16* __restrict__ p0, int ld, int lane) {
  const bf16* p = p0 + (lane & 15) * ld + ((lane & 16) ? 8 : 0);
  v16bf b;
#pragma unroll
  for (int t = 0; t < 8; ++t) b[t] = p[t];
#pragma unroll
  for (int t = 0; t < 8; ++t) b[8 + t] = p[16 + t];
  return b;
}

// ---------------------------------------------------------------------------
// Kernel 1: Wt[n][k] = bf16(W[k][n])   (128 x 512)
// ---------------------------------------------------------------------------
__global__ void k_prep_wt(const float* __restrict__ w, bf16* __restrict__ wt) {
  int idx = blockIdx.x * blockDim.x + threadIdx.x;      // over DIN*DOUT
  int k = idx / DOUT, n = idx % DOUT;
  wt[n * DIN + k] = (bf16)w[idx];
}

// ---------------------------------------------------------------------------
// Kernel 2: wh = x @ W   (per batch: 4096x512 @ 512x128)
// One wave -> 16 rows x 64 cols (4 accumulator tiles). 2x-unrolled ping-pong
// pipeline: loads for the next half-step land directly in the buffer that
// will be consumed next (no register copies).
// ---------------------------------------------------------------------------
__global__ void k_gemm_xw(const float* __restrict__ x, const bf16* __restrict__ wt,
                          float* __restrict__ wh, bf16* __restrict__ whT) {
  int w    = blockIdx.x * (blockDim.x >> 5) + (threadIdx.x >> 5);
  int lane = threadIdx.x & 31;
  int b      = w / ((NN / 16) * 2);
  int rem    = w % ((NN / 16) * 2);
  int m0     = (rem >> 1) * 16;
  int n_base = (rem & 1) * 64;

  const float* xb  = x + b * NN * DIN;
  const float* arow = xb + m0 * DIN;

  v8f acc[4] = {};
  v16bf aA = load_a_f32(arow, DIN, lane);
  v16bf bA[4];
#pragma unroll
  for (int nt = 0; nt < 4; ++nt)
    bA[nt] = load_b_T(wt + (n_base + nt * 16) * DIN, DIN, lane);

  for (int k0 = 0; k0 < DIN; k0 += 64) {
    int k1 = k0 + 32;                                 // DIN % 64 == 0 -> in range
    v16bf aB = load_a_f32(arow + k1, DIN, lane);
    v16bf bB[4];
#pragma unroll
    for (int nt = 0; nt < 4; ++nt)
      bB[nt] = load_b_T(wt + (n_base + nt * 16) * DIN + k1, DIN, lane);
#pragma unroll
    for (int nt = 0; nt < 4; ++nt)
      acc[nt] = wmma_bf16(aA, bA[nt], acc[nt]);

    int k2 = (k0 + 64 < DIN) ? k0 + 64 : 0;           // branch-free wrap
    aA = load_a_f32(arow + k2, DIN, lane);
#pragma unroll
    for (int nt = 0; nt < 4; ++nt)
      bA[nt] = load_b_T(wt + (n_base + nt * 16) * DIN + k2, DIN, lane);
#pragma unroll
    for (int nt = 0; nt < 4; ++nt)
      acc[nt] = wmma_bf16(aB, bB[nt], acc[nt]);
  }

  float* whb  = wh  + b * NN * DOUT;
  bf16*  whTb = whT + b * DOUT * NN;
  int mb = m0 + ((lane & 16) ? 8 : 0);
#pragma unroll
  for (int nt = 0; nt < 4; ++nt) {
    int n = n_base + nt * 16 + (lane & 15);
#pragma unroll
    for (int r = 0; r < 8; ++r) {
      float v = acc[nt][r];
      whb[(mb + r) * DOUT + n] = v;
      whTb[n * NN + (mb + r)]  = (bf16)v;     // 8 contiguous bf16 per lane
    }
  }
}

// ---------------------------------------------------------------------------
// Kernel 3: s_i = wh[i] . a_src ; d_i = wh[i] . a_dst   (one wave per row)
// ---------------------------------------------------------------------------
__global__ void k_row_sd(const float* __restrict__ wh, const float* __restrict__ a,
                         float* __restrict__ s, float* __restrict__ d) {
  int w    = blockIdx.x * (blockDim.x >> 5) + (threadIdx.x >> 5);   // NB*NN rows
  int lane = threadIdx.x & 31;
  const float* row = wh + w * DOUT;
  float sv = 0.f, dv = 0.f;
#pragma unroll
  for (int t = 0; t < 4; ++t) {
    int c = lane + t * 32;
    float v = row[c];
    sv += v * a[c];
    dv += v * a[DOUT + c];
  }
#pragma unroll
  for (int off = 16; off > 0; off >>= 1) {
    sv += __shfl_xor(sv, off, 32);
    dv += __shfl_xor(dv, off, 32);
  }
  if (lane == 0) { s[w] = sv; d[w] = dv; }
}

// ---------------------------------------------------------------------------
// Kernel 4: online-softmax row stats over masked leaky(s_i + d_j).
// Masked entries contribute e = NEG_INF; an all-masked row degenerates to the
// reference's uniform softmax automatically (exp(0)=1 per entry).
// ---------------------------------------------------------------------------
__global__ void k_stats(const float* __restrict__ adj, const float* __restrict__ s,
                        const float* __restrict__ d, float* __restrict__ mrow,
                        float* __restrict__ rsum) {
  int w    = blockIdx.x * (blockDim.x >> 5) + (threadIdx.x >> 5);   // NB*NN rows
  int lane = threadIdx.x & 31;
  int b = w >> 12, i = w & (NN - 1);
  float si = s[w];
  const float* arow = adj + i * NN;
  const float* db   = d + (b << 12);
  float m = NEG_INF, sum = 0.f;
  for (int j = lane; j < NN; j += 32) {
    if (j + 512 < NN) __builtin_prefetch(&arow[j + 512], 0, 0);  // global_prefetch_b8
    float e;
    if (arow[j] > 0.f) {
      e = si + db[j];
      e = e > 0.f ? e : 0.01f * e;
    } else {
      e = NEG_INF;
    }
    float nm = fmaxf(m, e);
    sum = sum * __expf(m - nm) + __expf(e - nm);
    m = nm;
  }
#pragma unroll
  for (int off = 16; off > 0; off >>= 1) {
    float om = __shfl_xor(m, off, 32);
    float os = __shfl_xor(sum, off, 32);
    float nm = fmaxf(m, om);
    sum = sum * __expf(m - nm) + os * __expf(om - nm);
    m = nm;
  }
  if (lane == 0) { mrow[w] = m; rsum[w] = sum; }
}

// ---------------------------------------------------------------------------
// Attention A-tile generator: unnormalized probabilities for 16x32 tile at
// (rows m0..m0+15, cols j0..j0+31), in WMMA A-fragment layout.
// ---------------------------------------------------------------------------
__device__ __forceinline__ v16bf make_p(const float* __restrict__ arow,
                                        const float* __restrict__ db,
                                        float si, float mi, int j0, int lane) {
  int jb = j0 + ((lane & 16) ? 8 : 0);
  v16bf p;
#pragma unroll
  for (int half = 0; half < 2; ++half) {
    int js = jb + half * 16;
#pragma unroll
    for (int t = 0; t < 8; ++t) {
      int   j = js + t;
      float e = si + db[j];
      e = e > 0.f ? e : 0.01f * e;
      e = (arow[j] > 0.f) ? e : NEG_INF;
      p[half * 8 + t] = (bf16)__expf(e - mi);
    }
  }
  return p;
}

// ---------------------------------------------------------------------------
// Kernel 5 (fused): out = softmax(mask(leaky(sᵢ+dⱼ))) @ wh, attention tile
// generated in registers (exp co-executes with XDL WMMA), 2x-unrolled
// ping-pong pipeline; 1/Σ folded into epilogue. Writes outT (bf16 [DOUT][NN]).
// ---------------------------------------------------------------------------
__global__ void k_att_wh(const float* __restrict__ adj, const float* __restrict__ s,
                         const float* __restrict__ d, const float* __restrict__ mrow,
                         const float* __restrict__ rsum, const bf16* __restrict__ whT,
                         bf16* __restrict__ outT) {
  int w    = blockIdx.x * (blockDim.x >> 5) + (threadIdx.x >> 5);
  int lane = threadIdx.x & 31;
  int b      = w / ((NN / 16) * 2);
  int rem    = w % ((NN / 16) * 2);
  int m0     = (rem >> 1) * 16;
  int n_base = (rem & 1) * 64;

  int   i  = m0 + (lane & 15);
  float si = s[b * NN + i];
  float mi = mrow[b * NN + i];
  const float* arow = adj + i * NN;
  const float* db   = d + b * NN;
  const bf16*  whTb = whT + b * DOUT * NN;

  v8f acc[4] = {};
  v16bf pA = make_p(arow, db, si, mi, 0, lane);
  v16bf bA[4];
#pragma unroll
  for (int nt = 0; nt < 4; ++nt)
    bA[nt] = load_b_T(whTb + (n_base + nt * 16) * NN, NN, lane);

  for (int j0 = 0; j0 < NN; j0 += 64) {
    int j1 = j0 + 32;                                 // NN % 64 == 0 -> in range
    v16bf pB = make_p(arow, db, si, mi, j1, lane);
    v16bf bB[4];
#pragma unroll
    for (int nt = 0; nt < 4; ++nt)
      bB[nt] = load_b_T(whTb + (n_base + nt * 16) * NN + j1, NN, lane);
#pragma unroll
    for (int nt = 0; nt < 4; ++nt)
      acc[nt] = wmma_bf16(pA, bA[nt], acc[nt]);

    int j2 = (j0 + 64 < NN) ? j0 + 64 : 0;            // branch-free wrap
    pA = make_p(arow, db, si, mi, j2, lane);
#pragma unroll
    for (int nt = 0; nt < 4; ++nt)
      bA[nt] = load_b_T(whTb + (n_base + nt * 16) * NN + j2, NN, lane);
#pragma unroll
    for (int nt = 0; nt < 4; ++nt)
      acc[nt] = wmma_bf16(pB, bB[nt], acc[nt]);
  }

  bf16* outTb = outT + b * DOUT * NN;
  int mb = m0 + ((lane & 16) ? 8 : 0);
#pragma unroll
  for (int nt = 0; nt < 4; ++nt) {
    int n = n_base + nt * 16 + (lane & 15);
#pragma unroll
    for (int r = 0; r < 8; ++r) {
      float inv = 1.f / rsum[b * NN + mb + r];
      outTb[n * NN + (mb + r)] = (bf16)(acc[nt][r] * inv);
    }
  }
}

// ---------------------------------------------------------------------------
// Kernel 6: y = relu(adj @ out)  -> d_out (f32, [B][NN][DOUT]), ping-pong.
// ---------------------------------------------------------------------------
__global__ void k_adj_out(const float* __restrict__ adj, const bf16* __restrict__ outT,
                          float* __restrict__ y) {
  int w    = blockIdx.x * (blockDim.x >> 5) + (threadIdx.x >> 5);
  int lane = threadIdx.x & 31;
  int b      = w / ((NN / 16) * 2);
  int rem    = w % ((NN / 16) * 2);
  int m0     = (rem >> 1) * 16;
  int n_base = (rem & 1) * 64;

  const float* arow  = adj + m0 * NN;
  const bf16*  outTb = outT + b * DOUT * NN;

  v8f acc[4] = {};
  v16bf aA = load_a_f32(arow, NN, lane);              // 0/1 exact in bf16
  v16bf bA[4];
#pragma unroll
  for (int nt = 0; nt < 4; ++nt)
    bA[nt] = load_b_T(outTb + (n_base + nt * 16) * NN, NN, lane);

  for (int j0 = 0; j0 < NN; j0 += 64) {
    int j1 = j0 + 32;                                 // NN % 64 == 0 -> in range
    v16bf aB = load_a_f32(arow + j1, NN, lane);
    v16bf bB[4];
#pragma unroll
    for (int nt = 0; nt < 4; ++nt)
      bB[nt] = load_b_T(outTb + (n_base + nt * 16) * NN + j1, NN, lane);
#pragma unroll
    for (int nt = 0; nt < 4; ++nt)
      acc[nt] = wmma_bf16(aA, bA[nt], acc[nt]);

    int j2 = (j0 + 64 < NN) ? j0 + 64 : 0;            // branch-free wrap
    aA = load_a_f32(arow + j2, NN, lane);
#pragma unroll
    for (int nt = 0; nt < 4; ++nt)
      bA[nt] = load_b_T(outTb + (n_base + nt * 16) * NN + j2, NN, lane);
#pragma unroll
    for (int nt = 0; nt < 4; ++nt)
      acc[nt] = wmma_bf16(aB, bB[nt], acc[nt]);
  }

  float* yb = y + b * NN * DOUT;
  int mb = m0 + ((lane & 16) ? 8 : 0);
#pragma unroll
  for (int nt = 0; nt < 4; ++nt) {
    int n = n_base + nt * 16 + (lane & 15);
#pragma unroll
    for (int r = 0; r < 8; ++r) {
      yb[(mb + r) * DOUT + n] = fmaxf(acc[nt][r], 0.f);
    }
  }
}

// ---------------------------------------------------------------------------
extern "C" void kernel_launch(void* const* d_in, const int* in_sizes, int n_in,
                              void* d_out, int out_size, void* d_ws, size_t ws_size,
                              hipStream_t stream) {
  const float* x   = (const float*)d_in[0];   // (B, N, DIN)
  const float* adj = (const float*)d_in[1];   // (N, N)
  const float* wgt = (const float*)d_in[2];   // (DIN, DOUT)
  const float* a   = (const float*)d_in[3];   // (2*DOUT, 1)
  float* y = (float*)d_out;                   // (B, N, DOUT)

  char*  ws  = (char*)d_ws;
  size_t off = 0;
  auto alloc = [&](size_t bytes) {
    void* p = ws + off;
    off = (off + bytes + 255) & ~(size_t)255;
    return p;
  };
  bf16*  wt   = (bf16*)alloc((size_t)DOUT * DIN * sizeof(bf16));
  float* wh   = (float*)alloc((size_t)NB * NN * DOUT * sizeof(float));
  bf16*  whT  = (bf16*)alloc((size_t)NB * DOUT * NN * sizeof(bf16));
  float* s    = (float*)alloc((size_t)NB * NN * sizeof(float));
  float* d    = (float*)alloc((size_t)NB * NN * sizeof(float));
  float* mrow = (float*)alloc((size_t)NB * NN * sizeof(float));
  float* rsum = (float*)alloc((size_t)NB * NN * sizeof(float));
  bf16*  outT = (bf16*)alloc((size_t)NB * DOUT * NN * sizeof(bf16));

  const int gemm_waves  = NB * (NN / 16) * 2;      // 2048
  const int gemm_blocks = gemm_waves / 4;          // 4 waves / block (128 thr)
  const int row_blocks  = (NB * NN) / 8;           // 8 waves / block (256 thr)

  k_prep_wt<<<(DIN * DOUT) / 256, 256, 0, stream>>>(wgt, wt);
  k_gemm_xw<<<gemm_blocks, 128, 0, stream>>>(x, wt, wh, whT);
  k_row_sd <<<row_blocks, 256, 0, stream>>>(wh, a, s, d);
  k_stats  <<<row_blocks, 256, 0, stream>>>(adj, s, d, mrow, rsum);
  k_att_wh <<<gemm_blocks, 128, 0, stream>>>(adj, s, d, mrow, rsum, whT, outT);
  k_adj_out<<<gemm_blocks, 128, 0, stream>>>(adj, outT, y);
}
